// QuantumDataTransformer_11630771437664
// MI455X (gfx1250) — matrix operand order repaired and checked by reading, not compiled, not run on hardware
//
#include <hip/hip_runtime.h>
#include <math.h>

// ---------------------------------------------------------------------------
// Problem constants (from reference)
// ---------------------------------------------------------------------------
#define BATCH 512
#define CH    65536      // 2^16 channels
#define HID   256
#define NQ    16

typedef __attribute__((ext_vector_type(16))) _Float16 v16h;
typedef __attribute__((ext_vector_type(8)))  _Float16 v8h;
typedef __attribute__((ext_vector_type(2)))  _Float16 v2h;
typedef __attribute__((ext_vector_type(8)))  float    v8f;
typedef __attribute__((ext_vector_type(4)))  float    v4f;
typedef __attribute__((ext_vector_type(2)))  float    v2f;

// D = A*B + C, f16 inputs, f32 accumulate (v_wmma_f32_16x16x32_f16)
__device__ __forceinline__ v8f wmma16x16x32(v16h a, v16h b, v8f c) {
  return __builtin_amdgcn_wmma_f32_16x16x32_f16(
      /*neg_a=*/false, a, /*neg_b=*/false, b,
      /*c_mod=*/(short)0, c, /*reuse_a=*/false, /*reuse_b=*/false);
}

// A-fragment: lane(half=lane>>4, m=lane&15) needs A[m][half*16 .. +15]
// -> two contiguous 16B chunks (ptr is 16B aligned).
__device__ __forceinline__ v16h frag_a(const _Float16* p) {
  v8h lo = *(const v8h*)(p);
  v8h hi = *(const v8h*)(p + 8);
  v16h r;
#pragma unroll
  for (int e = 0; e < 8; ++e) { r[e] = lo[e]; r[e + 8] = hi[e]; }
  return r;
}

// B-fragment from N-major LDS tile: lane needs Bt[n][half*16 .. +15],
// 16 contiguous f16 (two aligned 16B chunks).
__device__ __forceinline__ v16h frag_b(const _Float16* p) {
  v8h lo = *(const v8h*)(p);
  v8h hi = *(const v8h*)(p + 8);
  v16h r;
#pragma unroll
  for (int e = 0; e < 8; ++e) { r[e] = lo[e]; r[e + 8] = hi[e]; }
  return r;
}

// CDNA5 async global->LDS copy, 16 bytes, tracked by ASYNCcnt.
// lds = LDS byte address (low 32 bits of flat pointer), gp = global address.
__device__ __forceinline__ void async_g2l_b128(unsigned lds, const void* gp) {
  asm volatile("global_load_async_to_lds_b128 %0, %1, off"
               :: "v"(lds), "v"(gp) : "memory");
}
__device__ __forceinline__ void wait_asynccnt0() {
  asm volatile("s_wait_asynccnt 0" ::: "memory");
}

// ---------------------------------------------------------------------------
// Kernel 0: quantum angle tables  qtab[s*32 + 2i] = cos, [..+1] = sin,
//           and h1acc initialized with bias b1 (split-K GEMM1 adds into it).
// ---------------------------------------------------------------------------
__global__ __launch_bounds__(256) void prep_kernel(
    const float* __restrict__ x, const float* __restrict__ b1,
    float* __restrict__ h1acc, float* __restrict__ qtab) {
  int tid = blockIdx.x * 256 + threadIdx.x;
  h1acc[tid] = b1[tid & (HID - 1)];
  if (tid < BATCH * NQ) {
    int s = tid >> 4, i = tid & 15;
    float ang = 0.5f * 3.14159265358979323846f * x[(size_t)s * CH + i];
    qtab[2 * tid]     = cosf(ang);
    qtab[2 * tid + 1] = sinf(ang);
  }
}

// ---------------------------------------------------------------------------
// Kernel 1: GEMM1, h1acc += x[512,65536] @ W1[65536,256], split-K.
// grid (8 m-blocks, 64 k-splits), 256 threads (8 waves).
// WG tile: M=64, N=256, K-chunk=1024. Wave (msub = wave/2, nhalf = wave%2)
// owns a 16x128 strip -> 8 accumulators.
// Staging: A global_load_b128 (f32) -> cvt -> ds_store_b128; B scalar f32 loads -> cvt -> ds_store_b128.
// ---------------------------------------------------------------------------
#define G1_PA 40   // LDS pitch (halfwords): 16B-aligned frags, conflict-free A
__global__ __launch_bounds__(256) void gemm1_kernel(
    const float* __restrict__ x, const float* __restrict__ W1,
    float* __restrict__ h1acc) {
  __shared__ alignas(16) _Float16 As[64 * G1_PA];
  __shared__ alignas(16) _Float16 Bs[256 * G1_PA];
  const int tid  = threadIdx.x;
  const int lane = tid & 31;
  const int wave = tid >> 5;
  const int half = lane >> 4;
  const int l16  = lane & 15;
  const int m0 = blockIdx.x * 64;
  const int k0 = blockIdx.y * 1024;
  const int msub  = wave >> 1;          // 0..3
  const int nhalf = (wave & 1) * 128;   // 0 or 128
  const int am   = tid >> 2;            // A staging: row 0..63
  const int akk  = (tid & 3) * 8;       // A staging: 8-wide k segment

  const v8f zero = {0.f, 0.f, 0.f, 0.f, 0.f, 0.f, 0.f, 0.f};
  v8f acc[8];
#pragma unroll
  for (int t = 0; t < 8; ++t) acc[t] = zero;

  for (int kc = 0; kc < 1024; kc += 32) {
    // Stage A: x[m0..m0+63][k0+kc..+31] -> f16; 8 elems/thread,
    // two global b128 loads + one ds b128 store.
    {
      const float* gp = x + (size_t)(m0 + am) * CH + k0 + kc + akk;
      v4f d0 = *(const v4f*)(gp);
      v4f d1 = *(const v4f*)(gp + 4);
      v8h p;
#pragma unroll
      for (int e = 0; e < 4; ++e) {
        p[e]     = (_Float16)d0[e];
        p[e + 4] = (_Float16)d1[e];
      }
      *(v8h*)(As + am * G1_PA + akk) = p;
    }
    // Stage B transposed: W1[k][n] -> Bs[n][k]; thread t owns column n=t,
    // coalesced row reads, 4 packed ds b128 stores.
#pragma unroll
    for (int i8 = 0; i8 < 32; i8 += 8) {
      v8h p;
#pragma unroll
      for (int j = 0; j < 8; ++j)
        p[j] = (_Float16)W1[(size_t)(k0 + kc + i8 + j) * HID + tid];
      *(v8h*)(Bs + tid * G1_PA + i8) = p;
    }
    __syncthreads();

    v16h a = frag_a(As + (msub * 16 + l16) * G1_PA + half * 16);
#pragma unroll
    for (int t = 0; t < 8; ++t) {
      int n = nhalf + t * 16 + l16;
      v16h b = frag_b(Bs + n * G1_PA + half * 16);
      acc[t] = wmma16x16x32(a, b, acc[t]);
    }
    __syncthreads();
  }

  // Split-K reduction: atomic f32 add into h1acc
#pragma unroll
  for (int t = 0; t < 8; ++t) {
    int n = nhalf + t * 16 + l16;
#pragma unroll
    for (int v = 0; v < 8; ++v) {
      int m = m0 + msub * 16 + v + half * 8;
      atomicAdd(&h1acc[m * HID + n], acc[t][v]);
    }
  }
}

// ---------------------------------------------------------------------------
// Kernel 2: ReLU(h1acc) -> f16 (bias was pre-loaded by prep_kernel)
// ---------------------------------------------------------------------------
__global__ __launch_bounds__(256) void relu1_kernel(
    const float* __restrict__ h1acc, _Float16* __restrict__ h1h) {
  int tid = blockIdx.x * 256 + threadIdx.x;
  h1h[tid] = (_Float16)fmaxf(h1acc[tid], 0.f);
}

// ---------------------------------------------------------------------------
// Kernel 3: GEMM2, h2 = relu(h1 @ W2 + b2) -> f16.  M=512,N=256,K=256 (tiny).
// grid 32, 256 threads; WG tile M=16,N=256; wave owns two 16-col tiles.
// ---------------------------------------------------------------------------
__global__ __launch_bounds__(256) void gemm2_kernel(
    const _Float16* __restrict__ h1h, const float* __restrict__ W2,
    const float* __restrict__ b2, _Float16* __restrict__ h2h) {
  __shared__ alignas(16) _Float16 As[16 * G1_PA];
  const int tid  = threadIdx.x;
  const int lane = tid & 31;
  const int wave = tid >> 5;
  const int half = lane >> 4;
  const int l16  = lane & 15;
  const int m0   = blockIdx.x * 16;
  const int n0w  = wave * 32;

  const v8f zero = {0.f, 0.f, 0.f, 0.f, 0.f, 0.f, 0.f, 0.f};
  v8f acc[2] = {zero, zero};

  for (int kc = 0; kc < HID; kc += 32) {
    {
      // 512 halfwords = 256 dwords; one packed copy per thread
      int m = tid >> 4, c = (tid & 15) * 2;
      *(v2h*)(As + m * G1_PA + c) =
          *(const v2h*)(h1h + (m0 + m) * HID + kc + c);
    }
    __syncthreads();
    v16h a = frag_a(As + l16 * G1_PA + half * 16);
#pragma unroll
    for (int t = 0; t < 2; ++t) {
      int n = n0w + t * 16 + l16;
      v16h b;
#pragma unroll
      for (int e = 0; e < 16; ++e)
        b[e] = (_Float16)W2[(size_t)(kc + half * 16 + e) * HID + n];
      acc[t] = wmma16x16x32(a, b, acc[t]);
    }
    __syncthreads();
  }
#pragma unroll
  for (int t = 0; t < 2; ++t) {
    int n = n0w + t * 16 + l16;
    float bias = b2[n];
#pragma unroll
    for (int v = 0; v < 8; ++v) {
      int m = m0 + v + half * 8;
      h2h[m * HID + n] = (_Float16)fmaxf(acc[t][v] + bias, 0.f);
    }
  }
}

// ---------------------------------------------------------------------------
// Kernel 4: GEMM3 + fused quantum + combine.
// out[s][n] = 0.5f * ((h2 @ W3)[s][n] + b3[n] + Q[s][n])
// Q[s][n] = prod_{i=0..15} M_i[b_i, b_{i-1}],  M=[[c,-s],[s,c]], b_{-1}=0,
//           b_i = (n >> (15-i)) & 1   (qubit i has stride 2^(15-i)).
// grid 512 (column blocks of 128), 256 threads (8 waves, 16 cols each).
// W3 B-fragments loaded + converted ONCE into registers (W3 read exactly 1x),
// A tile (h2, f16) staged per M-step with GLOBAL_LOAD_ASYNC_TO_LDS_B128
// (ASYNCcnt), quantum (c,s) table via LDS.
// ---------------------------------------------------------------------------
#define G3_PA 264   // 256 + 8 halfwords: 16B-aligned, conflict-free frag loads
__global__ __launch_bounds__(256) void gemm3_kernel(
    const _Float16* __restrict__ h2h, const float* __restrict__ W3,
    const float* __restrict__ b3, const float* __restrict__ qtab,
    float* __restrict__ out) {
  __shared__ alignas(16) _Float16 As[16 * G3_PA];
  __shared__ alignas(16) float Qs[16 * 2 * NQ];   // 16 samples x (c,s)[16]
  const int tid  = threadIdx.x;
  const int lane = tid & 31;
  const int wave = tid >> 5;
  const int half = lane >> 4;
  const int l16  = lane & 15;
  const int n    = blockIdx.x * 128 + wave * 16 + l16;  // this lane's column

  // Preload all 8 B fragments (K=256) of W3 for column n into registers.
  v16h bfrag[8];
#pragma unroll
  for (int ks = 0; ks < 8; ++ks) {
    v16h b;
#pragma unroll
    for (int e = 0; e < 16; ++e)
      b[e] = (_Float16)W3[(size_t)(ks * 32 + half * 16 + e) * CH + n];
    bfrag[ks] = b;
  }
  const float bias = b3[n];

  for (int mstep = 0; mstep < BATCH / 16; ++mstep) {
    const int m0 = mstep * 16;
    // Stage A tile (16x256 f16 = 512 x b128) via async global->LDS DMA:
    // 2 transfers per thread, each lane supplies LDS dest + global src.
#pragma unroll
    for (int j = 0; j < 2; ++j) {
      int id = tid + j * 256;         // 0..511
      int m  = id >> 5;               // 32 b128 chunks per 256-halfword row
      int c8 = (id & 31) * 8;         // halfword offset, 16B aligned
      unsigned lds = (unsigned)(size_t)(As + m * G3_PA + c8);
      async_g2l_b128(lds, h2h + (size_t)(m0 + m) * HID + c8);
    }
    // Stage quantum table for these 16 samples: 512 floats
    ((v2f*)Qs)[tid] = ((const v2f*)(qtab + m0 * 2 * NQ))[tid];
    wait_asynccnt0();                 // own wave's async copies done
    __syncthreads();                  // + everyone else's

    v8f acc = {0.f, 0.f, 0.f, 0.f, 0.f, 0.f, 0.f, 0.f};
#pragma unroll
    for (int ks = 0; ks < 8; ++ks) {
      v16h a = frag_a(As + l16 * G3_PA + ks * 32 + half * 16);
      acc = wmma16x16x32(a, bfrag[ks], acc);
    }

    // Epilogue: bias + quantum chain product + scale, one store per element
#pragma unroll
    for (int v = 0; v < 8; ++v) {
      const int ml = v + half * 8;               // local sample index 0..15
      const float* qp = Qs + ml * 2 * NQ;
      float q = 1.f;
      int prev = 0;
#pragma unroll
      for (int i = 0; i < NQ; ++i) {
        float cv = qp[2 * i];
        float sv = qp[2 * i + 1];
        int y = (n >> (15 - i)) & 1;
        float f = (y == prev) ? cv : (prev ? -sv : sv);
        q *= f;
        prev = y;
      }
      out[(size_t)(m0 + ml) * CH + n] = 0.5f * (acc[v] + bias + q);
    }
    __syncthreads();
  }
}

// ---------------------------------------------------------------------------
// Launch
// ---------------------------------------------------------------------------
extern "C" void kernel_launch(void* const* d_in, const int* in_sizes, int n_in,
                              void* d_out, int out_size, void* d_ws, size_t ws_size,
                              hipStream_t stream) {
  const float* x  = (const float*)d_in[0];
  const float* W1 = (const float*)d_in[1];
  const float* b1 = (const float*)d_in[2];
  const float* W2 = (const float*)d_in[3];
  const float* b2 = (const float*)d_in[4];
  const float* W3 = (const float*)d_in[5];
  const float* b3 = (const float*)d_in[6];
  float* out = (float*)d_out;

  // Workspace layout (~1.1 MB total)
  char* ws = (char*)d_ws;
  float*    h1acc = (float*)ws;                                  // 512*256*4
  _Float16* h1h   = (_Float16*)(ws + 512 * 1024);                // 512*256*2
  _Float16* h2h   = (_Float16*)(ws + 512 * 1024 + 256 * 1024);   // 512*256*2
  float*    qtab  = (float*)(ws + 1024 * 1024);                  // 512*32*4

  prep_kernel <<<BATCH, 256, 0, stream>>>(x, b1, h1acc, qtab);
  gemm1_kernel<<<dim3(BATCH / 64, 64), 256, 0, stream>>>(x, W1, h1acc);
  relu1_kernel<<<BATCH, 256, 0, stream>>>(h1acc, h1h);
  gemm2_kernel<<<BATCH / 16, 256, 0, stream>>>(h1h, W2, b2, h2h);
  gemm3_kernel<<<CH / 128, 256, 0, stream>>>(h2h, W3, b3, qtab, out);
}
